// VectorQuantizer_27650999452558
// MI455X (gfx1250) — matrix operand order repaired and checked by reading, compiled, not run on hardware
//
#include <hip/hip_runtime.h>

// ---------------------------------------------------------------------------
// CDNA5 (gfx1250) VQ-VAE vector quantizer.
//   d = ||z||^2 + ||e||^2 - 2 z@E^T  via v_wmma_f32_16x16x32_bf16
// ---------------------------------------------------------------------------

typedef __attribute__((ext_vector_type(16))) __bf16 v16bf;
typedef __attribute__((ext_vector_type(8)))  float  v8f;

union BF16x16 { uint4 q[2]; v16bf v; };

// Problem sizes (fixed by the reference)
constexpr int  kC    = 256;     // e_dim
constexpr int  kNE   = 1024;    // codebook size
constexpr long kNpts = 32768;   // 32*32*32 points
// d_out layout: z_q | loss | indices | d   (flat, return order)
constexpr long ZQ_CNT   = 8388608;          // 32*256*32*32
constexpr long LOSS_OFF = ZQ_CNT;           // 8388608
constexpr long IDX_OFF  = LOSS_OFF + 1;     // 8388609
constexpr long D_OFF    = IDX_OFF + kNpts;  // 8421377
constexpr float LOSS_SCALE = 1.25f / 8388608.0f;  // (1+beta)/numel

constexpr int PADC = 264;       // padded K stride for LDS A tile (bank-safe, 16B-aligned)

__device__ __forceinline__ unsigned short f2bf(float f) {
  unsigned int u = __float_as_uint(f);
  u += 0x7FFFu + ((u >> 16) & 1u);          // round-to-nearest-even
  return (unsigned short)(u >> 16);
}

// order-preserving float -> uint mapping (handles tiny negative d from rounding)
__device__ __forceinline__ unsigned int fkey(float x) {
  unsigned int u = __float_as_uint(x);
  return (u & 0x80000000u) ? ~u : (u | 0x80000000u);
}

// ---------------------------------------------------------------------------
// Kernel 1: codebook fp32 -> bf16, per-code squared norms
// ---------------------------------------------------------------------------
__global__ __launch_bounds__(256) void vq_prep(const float* __restrict__ emb,
                                               unsigned short* __restrict__ embB,
                                               float* __restrict__ e2) {
  __shared__ float red[256];
  const int code = blockIdx.x;
  const int t = threadIdx.x;
  const float v = emb[code * kC + t];
  embB[code * kC + t] = f2bf(v);
  red[t] = v * v;
  __syncthreads();
  for (int s = 128; s > 0; s >>= 1) {
    if (t < s) red[t] += red[t + s];
    __syncthreads();
  }
  if (t == 0) e2[code] = red[0];
}

// ---------------------------------------------------------------------------
// Kernel 2: per-block = 16 z rows x all 1024 codes.
//   8 waves, each wave: 16 rows x 128 codes, K=256 in 8 steps of 32 (bf16 WMMA)
// ---------------------------------------------------------------------------
__global__ __launch_bounds__(256) void vq_main(const float* __restrict__ z,
                                               const float* __restrict__ emb,
                                               const unsigned short* __restrict__ embB,
                                               const float* __restrict__ e2g,
                                               float* __restrict__ out,
                                               float* __restrict__ partials) {
  __shared__ unsigned short As[16 * PADC];   // 16 rows x 256 K (bf16), padded
  __shared__ float red[256];
  __shared__ float z2s[16];
  __shared__ unsigned long long best[16];
  __shared__ unsigned int idxs[16];

  const int t  = threadIdx.x;
  const int nb = blockIdx.x * 16;            // first flat point index of tile
  const int b  = nb >> 10;
  const int h  = (nb >> 5) & 31;
  const int wb = nb & 31;                    // 0 or 16 (16 | 32)
  // z[b][c][h][w] flat index = (b*256 + c)*1024 + h*32 + w
  const long zplane = (long)b * kC * 1024 + h * 32 + wb;

  if (t < 16) best[t] = ~0ull;
  __syncthreads();

  // ---- Phase 1: load 16x256 z tile (coalesced along w), bf16 into LDS ----
  {
    const int r  = t & 15;                   // row within tile (== w offset)
    const int cg = t >> 4;                   // c-pair group 0..15
    float ss = 0.f;
    for (int j = 0; j < 8; ++j) {
      const int c = j * 32 + cg * 2;
      const float v0 = z[zplane + (long)c * 1024 + r];
      const float v1 = z[zplane + (long)(c + 1) * 1024 + r];
      ss += v0 * v0 + v1 * v1;
      const unsigned int packed =
          (unsigned int)f2bf(v0) | ((unsigned int)f2bf(v1) << 16);
      *(unsigned int*)&As[r * PADC + c] = packed;
    }
    red[t] = ss;
  }
  __syncthreads();
  if (t < 16) {                              // deterministic fixed-order z^2 sum
    float s = 0.f;
    for (int j = 0; j < 16; ++j) s += red[j * 16 + t];
    z2s[t] = s;
  }
  __syncthreads();

  // ---- Phase 2: WMMA GEMM  acc[m][n] = sum_k zbf[m][k]*ebf[n][k] ----
  const int wave = t >> 5;
  const int lane = t & 31;
  const int half = lane >> 4;
  const int ln   = lane & 15;

  v8f acc[8];
  const v8f vzero = {0.f, 0.f, 0.f, 0.f, 0.f, 0.f, 0.f, 0.f};
#pragma unroll
  for (int f = 0; f < 8; ++f) acc[f] = vzero;

  const unsigned short* Arow = &As[ln * PADC];
  for (int kk = 0; kk < kC; kk += 32) {
    // A fragment (16x32 bf16): lane m=ln; K = half*16 + {0..15} contiguous
    BF16x16 ua;
    ua.q[0] = *(const uint4*)&Arow[kk + half * 16];
    ua.q[1] = *(const uint4*)&Arow[kk + half * 16 + 8];
#pragma unroll
    for (int f = 0; f < 8; ++f) {
      // B fragment (32x16 bf16): col n=ln; K = half*16 + {0..15} contiguous
      const int code = wave * 128 + f * 16 + ln;
      const uint4* pe = (const uint4*)&embB[code * kC + kk + half * 16];
      BF16x16 ub;
      ub.q[0] = pe[0];
      ub.q[1] = pe[1];
      acc[f] = __builtin_amdgcn_wmma_f32_16x16x32_bf16(
          false, ua.v, false, ub.v, (short)0, acc[f], false, false);
    }
  }

  // ---- Phase 3: d = z2 + e2 - 2*dot, store d, block-local argmin ----
  float* dout = out + D_OFF + (long)nb * 1024;
  unsigned long long mykey[8];
#pragma unroll
  for (int r = 0; r < 8; ++r) mykey[r] = ~0ull;
#pragma unroll
  for (int f = 0; f < 8; ++f) {
    const int code = wave * 128 + f * 16 + ln;
    const float e2v = e2g[code];
#pragma unroll
    for (int r = 0; r < 8; ++r) {
      const int row = r + half * 8;          // C/D layout: VGPR r, halves -> M
      const float dval = z2s[row] + e2v - 2.0f * acc[f][r];
      dout[(long)row * 1024 + code] = dval;
      const unsigned long long key =
          ((unsigned long long)fkey(dval) << 32) | (unsigned int)code;
      if (key < mykey[r]) mykey[r] = key;
    }
  }
#pragma unroll
  for (int r = 0; r < 8; ++r)
    atomicMin(&best[r + half * 8], mykey[r]);  // ds_min_u64; tie -> lowest index
  __syncthreads();

  // ---- Phase 4: indices ----
  if (t < 16) {
    const unsigned int idx = (unsigned int)best[t];
    idxs[t] = idx;
    out[IDX_OFF + nb + t] = (float)idx;
  }
  __syncthreads();

  // ---- Phase 5: z_q gather (fp32 codebook) + loss partial ----
  {
    const int r  = t & 15;
    const int cg = t >> 4;
    const float* erow = emb + (long)idxs[r] * kC;
    float ls = 0.f;
    for (int j = 0; j < 16; ++j) {
      const int c = cg + 16 * j;
      const float q = erow[c];
      const long a = zplane + (long)c * 1024 + r;   // BCHW address; ZQ_OFF==0
      const float dd = q - z[a];
      ls += dd * dd;
      out[a] = q;                            // straight-through value == z_q
    }
    red[t] = ls;
  }
  __syncthreads();
  for (int s = 128; s > 0; s >>= 1) {        // deterministic tree reduce
    if (t < s) red[t] += red[t + s];
    __syncthreads();
  }
  if (t == 0) partials[blockIdx.x] = red[0];
}

// ---------------------------------------------------------------------------
// Kernel 3: deterministic final loss reduction over 2048 block partials
// ---------------------------------------------------------------------------
__global__ __launch_bounds__(256) void vq_loss_reduce(const float* __restrict__ partials,
                                                      float* __restrict__ out) {
  __shared__ float red[256];
  const int t = threadIdx.x;
  float s = 0.f;
  for (int j = 0; j < 8; ++j) s += partials[t + 256 * j];
  red[t] = s;
  __syncthreads();
  for (int st = 128; st > 0; st >>= 1) {
    if (t < st) red[t] += red[t + st];
    __syncthreads();
  }
  if (t == 0) out[LOSS_OFF] = red[0] * LOSS_SCALE;
}

// ---------------------------------------------------------------------------
extern "C" void kernel_launch(void* const* d_in, const int* in_sizes, int n_in,
                              void* d_out, int out_size, void* d_ws, size_t ws_size,
                              hipStream_t stream) {
  const float* z   = (const float*)d_in[0];    // (32,256,32,32) fp32
  const float* emb = (const float*)d_in[1];    // (1024,256) fp32
  float* out = (float*)d_out;

  // workspace: bf16 codebook | e2 norms | per-block loss partials
  unsigned short* embB = (unsigned short*)d_ws;                       // 512 KB
  float* e2       = (float*)((char*)d_ws + (size_t)kNE * kC * 2);     //   4 KB
  float* partials = e2 + kNE;                                         //   8 KB

  vq_prep<<<kNE, 256, 0, stream>>>(emb, embB, e2);
  vq_main<<<(int)(kNpts / 16), 256, 0, stream>>>(z, emb, embB, e2, out, partials);
  vq_loss_reduce<<<1, 256, 0, stream>>>(partials, out);
}
